// MotionEncoder_39960375722296
// MI455X (gfx1250) — matrix operand order repaired and checked by reading, not compiled, not run on hardware
//
#include <hip/hip_runtime.h>
#include <hip/hip_bf16.h>
#include <stdint.h>
#include <stddef.h>

// ---------------------------------------------------------------------------
// Types for CDNA5 WMMA (wave32): v_wmma_f32_16x16x32_bf16
// ---------------------------------------------------------------------------
typedef __bf16 bf16_t;
typedef __attribute__((ext_vector_type(16))) __bf16 bf16x16;
typedef __attribute__((ext_vector_type(8)))  __bf16 bf16x8;
typedef __attribute__((ext_vector_type(8)))  float  f32x8;

__device__ __forceinline__ bf16_t f2bf(float f) {
  union { float f; uint32_t u; } v; v.f = f;
  uint32_t r = v.u + 0x7FFFu + ((v.u >> 16) & 1u);   // round-to-nearest-even
  uint16_t h = (uint16_t)(r >> 16);
  union { uint16_t u; bf16_t b; } o; o.u = h;
  return o.b;
}

__device__ __forceinline__ float gelu_exact(float x) {
  return 0.5f * x * (1.0f + erff(x * 0.70710678118654752440f));
}

// A tile (16x32 MxK, bf16), ISA layout: lanes 0-15 row M=lane, K in
// {k0..k0+7, k0+16..k0+23}; lanes 16-31 same rows, K in {k0+8..15, k0+24..31}.
__device__ __forceinline__ bf16x16 load_a_tile(const bf16_t* __restrict__ base,
                                               long ld, int m0, int k0, int lane) {
  int r = m0 + (lane & 15);
  int sel = (lane >> 4) & 1;
  const bf16_t* p = base + (long)r * ld + k0 + sel * 8;
  bf16x8 lo = *(const bf16x8*)(p);
  bf16x8 hi = *(const bf16x8*)(p + 16);
  return __builtin_shufflevector(lo, hi, 0,1,2,3,4,5,6,7,8,9,10,11,12,13,14,15);
}

// B tile (32x16 KxN, bf16) from row-major weights [N,K]: lane holds column
// N=lane&15; lanes 0-15 K in {k0..k0+7, k0+16..k0+23}; lanes 16-31 K in {k0+8..15, k0+24..31}.
__device__ __forceinline__ bf16x16 load_b_tile(const bf16_t* __restrict__ base,
                                               long ld, int n0, int k0, int lane) {
  int n = n0 + (lane & 15);
  int sel = (lane >> 4) & 1;
  const bf16_t* p = base + (long)n * ld + k0 + sel * 8;
  bf16x8 lo = *(const bf16x8*)(p);
  bf16x8 hi = *(const bf16x8*)(p + 16);
  return __builtin_shufflevector(lo, hi, 0,1,2,3,4,5,6,7,8,9,10,11,12,13,14,15);
}

__device__ __forceinline__ f32x8 wmma_bf16(bf16x16 a, bf16x16 b, f32x8 c) {
  return __builtin_amdgcn_wmma_f32_16x16x32_bf16(false, a, false, b,
                                                 (short)0, c, false, false);
}

// ---------------------------------------------------------------------------
// f32 -> bf16 conversion
// ---------------------------------------------------------------------------
__global__ __launch_bounds__(256)
void f32_to_bf16_kernel(const float* __restrict__ src, bf16_t* __restrict__ dst,
                        long n) {
  long i = (long)blockIdx.x * 256 + threadIdx.x;
  if (i < n) dst[i] = f2bf(src[i]);
}

// ---------------------------------------------------------------------------
// Register-blocked GEMM: C[M,N](f32) = A[M,K](bf16) * W[N,K]^T(bf16) (+biases)
// Block = 256 threads = 8 waves. Each wave computes a 32x64 macro-tile
// (2 M-tiles x 4 N-tiles = 8 WMMA accumulators). Per K-step: 2 A loads +
// 4 B loads feed 8 WMMAs. Block tile = 32(M) x 512(N).
// grid = (N/512, M/32). Requires M%32==0, N%512==0, K%32==0.
// ---------------------------------------------------------------------------
__global__ __launch_bounds__(256)
void gemm_bf16_kernel(const bf16_t* __restrict__ A, long lda,
                      const bf16_t* __restrict__ W, long ldw,
                      float* __restrict__ C, long ldc, int K,
                      const float* __restrict__ bias0,
                      const float* __restrict__ bias1) {
  const int lane = threadIdx.x & 31;
  const int wave = threadIdx.x >> 5;
  const int n0 = blockIdx.x * 512 + wave * 64;
  const int m0 = blockIdx.y * 32;

  f32x8 acc[2][4];
#pragma unroll
  for (int i = 0; i < 2; ++i)
#pragma unroll
    for (int j = 0; j < 4; ++j)
      acc[i][j] = (f32x8){0.f, 0.f, 0.f, 0.f, 0.f, 0.f, 0.f, 0.f};

  for (int k0 = 0; k0 < K; k0 += 32) {
    bf16x16 a0 = load_a_tile(A, lda, m0, k0, lane);
    bf16x16 a1 = load_a_tile(A, lda, m0 + 16, k0, lane);
    bf16x16 b0 = load_b_tile(W, ldw, n0 + 0,  k0, lane);
    bf16x16 b1 = load_b_tile(W, ldw, n0 + 16, k0, lane);
    bf16x16 b2 = load_b_tile(W, ldw, n0 + 32, k0, lane);
    bf16x16 b3 = load_b_tile(W, ldw, n0 + 48, k0, lane);
    acc[0][0] = wmma_bf16(a0, b0, acc[0][0]);
    acc[1][0] = wmma_bf16(a1, b0, acc[1][0]);
    acc[0][1] = wmma_bf16(a0, b1, acc[0][1]);
    acc[1][1] = wmma_bf16(a1, b1, acc[1][1]);
    acc[0][2] = wmma_bf16(a0, b2, acc[0][2]);
    acc[1][2] = wmma_bf16(a1, b2, acc[1][2]);
    acc[0][3] = wmma_bf16(a0, b3, acc[0][3]);
    acc[1][3] = wmma_bf16(a1, b3, acc[1][3]);
  }

#pragma unroll
  for (int j = 0; j < 4; ++j) {
    const int n = n0 + j * 16 + (lane & 15);
    float bias = 0.0f;
    if (bias0) bias += bias0[n];
    if (bias1) bias += bias1[n];
#pragma unroll
    for (int i = 0; i < 2; ++i) {
      const int mrow = m0 + i * 16 + ((lane >> 4) << 3);
#pragma unroll
      for (int v = 0; v < 8; ++v)
        C[(long)(mrow + v) * ldc + n] = acc[i][j][v] + bias;
    }
  }
}

// ---------------------------------------------------------------------------
// One LSTM timestep (B=32): g = xg[:,t,:] + h_{t-1} * W_hh^T, then cell update.
// grid = H/32 blocks; block = 256 thr = 8 waves.
// Wave w: gate = w>>1, n-subtile = w&1; computes BOTH M-tiles sharing one
// B load per K-step (2 WMMAs / 3 tile loads). Gates staged through 16 KB LDS,
// then fused sigmoid/tanh cell update.
// ---------------------------------------------------------------------------
__global__ __launch_bounds__(256)
void lstm_step_kernel(const float* __restrict__ xg,       // [B*T, 4H]
                      const bf16_t* __restrict__ hprev,   // row base for t-1 (ld=T*H), null at t=0
                      const bf16_t* __restrict__ Whh,     // [4H, H] bf16
                      float* __restrict__ c_state,        // [B, H]
                      bf16_t* __restrict__ hseq_bf,       // [B, T, H]
                      float* __restrict__ hseq_f32,       // [B, T, H] or null
                      int t, int T, int H) {
  __shared__ float gs[4][32][32];
  const int lane = threadIdx.x & 31;
  const int wave = threadIdx.x >> 5;
  const int n0 = blockIdx.x * 32;

  if (hprev) {
    const int gate = wave >> 1;
    const int ns = wave & 1;
    const int ncol = n0 + ns * 16;
    const long ldA = (long)T * H;
    const int wrow0 = gate * H + ncol;
    f32x8 acc0 = {0.f, 0.f, 0.f, 0.f, 0.f, 0.f, 0.f, 0.f};
    f32x8 acc1 = {0.f, 0.f, 0.f, 0.f, 0.f, 0.f, 0.f, 0.f};
    for (int k0 = 0; k0 < H; k0 += 32) {
      bf16x16 b  = load_b_tile(Whh, H, wrow0, k0, lane);
      bf16x16 a0 = load_a_tile(hprev, ldA, 0,  k0, lane);
      bf16x16 a1 = load_a_tile(hprev, ldA, 16, k0, lane);
      acc0 = wmma_bf16(a0, b, acc0);
      acc1 = wmma_bf16(a1, b, acc1);
    }
    const int n = ns * 16 + (lane & 15);
    const int mr = (lane >> 4) << 3;
#pragma unroll
    for (int v = 0; v < 8; ++v) gs[gate][mr + v][n] = acc0[v];
#pragma unroll
    for (int v = 0; v < 8; ++v) gs[gate][16 + mr + v][n] = acc1[v];
  } else {
    for (int i = threadIdx.x; i < 4 * 32 * 32; i += 256)
      ((float*)gs)[i] = 0.0f;
  }
  __syncthreads();

  // Pointwise cell update: 32 batch rows x 32 columns = 1024 elements.
  for (int idx = threadIdx.x; idx < 32 * 32; idx += 256) {
    const int b = idx >> 5;
    const int col = idx & 31;
    const int hcol = n0 + col;
    const long row = (long)b * T + t;
    const long xbase = row * (4L * H);
    float gi = xg[xbase + 0 * H + hcol] + gs[0][b][col];
    float gf = xg[xbase + 1 * H + hcol] + gs[1][b][col];
    float gg = xg[xbase + 2 * H + hcol] + gs[2][b][col];
    float go = xg[xbase + 3 * H + hcol] + gs[3][b][col];
    float i_ = 1.0f / (1.0f + __expf(-gi));
    float f_ = 1.0f / (1.0f + __expf(-gf));
    float g_ = tanhf(gg);
    float o_ = 1.0f / (1.0f + __expf(-go));
    const long cidx = (long)b * H + hcol;
    float cprev = hprev ? c_state[cidx] : 0.0f;   // t==0: fresh state, no read
    float c = f_ * cprev + i_ * g_;
    c_state[cidx] = c;
    float h = o_ * tanhf(c);
    const long hidx = row * H + hcol;
    hseq_bf[hidx] = f2bf(h);
    if (hseq_f32) hseq_f32[hidx] = h;
  }
}

// ---------------------------------------------------------------------------
// LayerNorm -> exact GELU -> bf16 (one block per (b,t) row, H=1024)
// ---------------------------------------------------------------------------
__global__ __launch_bounds__(256)
void ln_gelu_bf16_kernel(const float* __restrict__ X, const float* __restrict__ w,
                         const float* __restrict__ bi, bf16_t* __restrict__ Y,
                         int H) {
  const long row = blockIdx.x;
  const float* x = X + row * (long)H;
  __shared__ float sh_s[256], sh_q[256];
  float s = 0.0f, q = 0.0f;
  for (int i = threadIdx.x; i < H; i += 256) { float v = x[i]; s += v; q += v * v; }
  sh_s[threadIdx.x] = s; sh_q[threadIdx.x] = q;
  __syncthreads();
  for (int st = 128; st > 0; st >>= 1) {
    if (threadIdx.x < st) {
      sh_s[threadIdx.x] += sh_s[threadIdx.x + st];
      sh_q[threadIdx.x] += sh_q[threadIdx.x + st];
    }
    __syncthreads();
  }
  const float mu = sh_s[0] / H;
  const float var = sh_q[0] / H - mu * mu;
  const float rstd = rsqrtf(var + 1e-5f);
  for (int i = threadIdx.x; i < H; i += 256) {
    float v = (x[i] - mu) * rstd * w[i] + bi[i];
    Y[row * (long)H + i] = f2bf(gelu_exact(v));
  }
}

// ---------------------------------------------------------------------------
// exact GELU -> LayerNorm -> f32 out (one block per row, H=1024 exactly)
// ---------------------------------------------------------------------------
__global__ __launch_bounds__(256)
void gelu_ln_out_kernel(const float* __restrict__ X, const float* __restrict__ w,
                        const float* __restrict__ bi, float* __restrict__ Y,
                        int H) {
  const long row = blockIdx.x;
  const float* x = X + row * (long)H;
  float vals[4];
  __shared__ float sh_s[256], sh_q[256];
  float s = 0.0f, q = 0.0f;
#pragma unroll
  for (int j = 0; j < 4; ++j) {
    int i = threadIdx.x + j * 256;
    float v = gelu_exact(x[i]);
    vals[j] = v; s += v; q += v * v;
  }
  sh_s[threadIdx.x] = s; sh_q[threadIdx.x] = q;
  __syncthreads();
  for (int st = 128; st > 0; st >>= 1) {
    if (threadIdx.x < st) {
      sh_s[threadIdx.x] += sh_s[threadIdx.x + st];
      sh_q[threadIdx.x] += sh_q[threadIdx.x + st];
    }
    __syncthreads();
  }
  const float mu = sh_s[0] / H;
  const float var = sh_q[0] / H - mu * mu;
  const float rstd = rsqrtf(var + 1e-5f);
#pragma unroll
  for (int j = 0; j < 4; ++j) {
    int i = threadIdx.x + j * 256;
    Y[row * (long)H + i] = (vals[j] - mu) * rstd * w[i] + bi[i];
  }
}

// ---------------------------------------------------------------------------
// Host-side orchestration
// ---------------------------------------------------------------------------
extern "C" void kernel_launch(void* const* d_in, const int* in_sizes, int n_in,
                              void* d_out, int out_size, void* d_ws, size_t ws_size,
                              hipStream_t stream) {
  (void)in_sizes; (void)n_in; (void)out_size; (void)ws_size;
  const int B = 32, T = 512, D = 256, H = 1024;

  const float* x_f32   = (const float*)d_in[0];
  const float* W_ih0   = (const float*)d_in[1];
  const float* W_hh0   = (const float*)d_in[2];
  const float* b_ih0   = (const float*)d_in[3];
  const float* b_hh0   = (const float*)d_in[4];
  const float* W_ih1   = (const float*)d_in[5];
  const float* W_hh1   = (const float*)d_in[6];
  const float* b_ih1   = (const float*)d_in[7];
  const float* b_hh1   = (const float*)d_in[8];
  const float* ln1_w   = (const float*)d_in[9];
  const float* ln1_b   = (const float*)d_in[10];
  const float* lin_w   = (const float*)d_in[11];
  const float* lin_b   = (const float*)d_in[12];
  const float* ln2_w   = (const float*)d_in[13];
  const float* ln2_b   = (const float*)d_in[14];

  // Workspace carve-out (256B aligned regions)
  char* wsp = (char*)d_ws;
  auto carve = [&wsp](size_t bytes) -> char* {
    char* p = wsp;
    wsp += (bytes + 255) & ~(size_t)255;
    return p;
  };
  const long BT = (long)B * T;
  bf16_t* x_bf     = (bf16_t*)carve((size_t)BT * D * 2);        // 8 MB
  bf16_t* Wih0b    = (bf16_t*)carve((size_t)4 * H * D * 2);     // 2 MB
  bf16_t* Whh0b    = (bf16_t*)carve((size_t)4 * H * H * 2);     // 8 MB
  bf16_t* Wih1b    = (bf16_t*)carve((size_t)4 * H * H * 2);     // 8 MB
  bf16_t* Whh1b    = (bf16_t*)carve((size_t)4 * H * H * 2);     // 8 MB
  bf16_t* linWb    = (bf16_t*)carve((size_t)H * H * 2);         // 2 MB
  bf16_t* h0seq_bf = (bf16_t*)carve((size_t)BT * H * 2);        // 32 MB (reused as act_bf)
  bf16_t* h1seq_bf = (bf16_t*)carve((size_t)BT * H * 2);        // 32 MB
  float*  h1seq_f  = (float*)carve((size_t)BT * H * 4);         // 64 MB
  float*  c_state  = (float*)carve((size_t)B * H * 4);          // 128 KB (shared by both layers)
  float*  xg       = (float*)carve((size_t)BT * 4 * H * 4);     // 256 MB (reused: xg0, xg1, lin_out)

  auto cvt = [&](const float* src, bf16_t* dst, long n) {
    f32_to_bf16_kernel<<<dim3((unsigned)((n + 255) / 256)), 256, 0, stream>>>(src, dst, n);
  };

  // 1) Convert inputs/weights to bf16
  cvt(x_f32, x_bf,  BT * (long)D);
  cvt(W_ih0, Wih0b, 4L * H * D);
  cvt(W_hh0, Whh0b, 4L * H * H);
  cvt(W_ih1, Wih1b, 4L * H * H);
  cvt(W_hh1, Whh1b, 4L * H * H);
  cvt(lin_w, linWb, (long)H * H);

  // 2) xg0 = x * W_ih0^T + b_ih0 + b_hh0   [B*T, 4H]
  gemm_bf16_kernel<<<dim3(4 * H / 512, (unsigned)(BT / 32)), 256, 0, stream>>>(
      x_bf, D, Wih0b, D, xg, 4 * H, D, b_ih0, b_hh0);

  // 3) Layer-0 recurrence
  for (int t = 0; t < T; ++t) {
    const bf16_t* hp = (t == 0) ? nullptr : (h0seq_bf + (long)(t - 1) * H);
    lstm_step_kernel<<<H / 32, 256, 0, stream>>>(
        xg, hp, Whh0b, c_state, h0seq_bf, nullptr, t, T, H);
  }

  // 4) xg1 = h0 * W_ih1^T + b_ih1 + b_hh1  (reuses xg buffer)
  gemm_bf16_kernel<<<dim3(4 * H / 512, (unsigned)(BT / 32)), 256, 0, stream>>>(
      h0seq_bf, H, Wih1b, H, xg, 4 * H, H, b_ih1, b_hh1);

  // 5) Layer-1 recurrence (also emits f32 h for LayerNorm)
  for (int t = 0; t < T; ++t) {
    const bf16_t* hp = (t == 0) ? nullptr : (h1seq_bf + (long)(t - 1) * H);
    lstm_step_kernel<<<H / 32, 256, 0, stream>>>(
        xg, hp, Whh1b, c_state, h1seq_bf, h1seq_f, t, T, H);
  }

  // 6) LN1 -> GELU -> bf16 activations (reuse h0seq_bf region)
  bf16_t* act_bf = h0seq_bf;
  ln_gelu_bf16_kernel<<<(unsigned)BT, 256, 0, stream>>>(h1seq_f, ln1_w, ln1_b, act_bf, H);

  // 7) Linear: lin_out = act * lin_w^T + lin_b  (reuse xg region as f32 [B*T, H])
  float* lin_out = xg;
  gemm_bf16_kernel<<<dim3(H / 512, (unsigned)(BT / 32)), 256, 0, stream>>>(
      act_bf, H, linWb, H, lin_out, H, H, lin_b, nullptr);

  // 8) GELU -> LN2 -> f32 output
  gelu_ln_out_kernel<<<(unsigned)BT, 256, 0, stream>>>(lin_out, ln2_w, ln2_b, (float*)d_out, H);
}
